// Head_35029753266855
// MI455X (gfx1250) — compile-verified, matrix operand order checked by reading, not run on hardware
//
#include <hip/hip_runtime.h>

typedef _Float16 half16 __attribute__((ext_vector_type(16)));
typedef _Float16 half8  __attribute__((ext_vector_type(8)));
typedef _Float16 half4  __attribute__((ext_vector_type(4)));
typedef float    f8v    __attribute__((ext_vector_type(8)));
typedef float    f4v    __attribute__((ext_vector_type(4)));
typedef int      v4i    __attribute__((ext_vector_type(4)));

#define NB 16
#define NT 2048
#define NC 1024
#define ND 128

typedef __attribute__((address_space(1))) v4i gas_v4i;  // global int4
typedef __attribute__((address_space(3))) v4i las_v4i;  // LDS int4

__device__ __forceinline__ half16 cat8(half8 a, half8 b) {
  return __builtin_shufflevector(a, b, 0,1,2,3,4,5,6,7,8,9,10,11,12,13,14,15);
}

__device__ __forceinline__ f8v wmma16(half16 a, half16 b, f8v c) {
  // D = A(16x32 f16) * B(32x16 f16) + C(16x16 f32)
  return __builtin_amdgcn_wmma_f32_16x16x32_f16(false, a, false, b, (short)0, c,
                                                false, false);
}

// 16-byte global -> LDS copy. Async (ASYNCcnt-tracked DMA) when available.
__device__ __forceinline__ void cp16_async(const _Float16* g, _Float16* l) {
#if __has_builtin(__builtin_amdgcn_global_load_async_to_lds_b128)
  __builtin_amdgcn_global_load_async_to_lds_b128((gas_v4i*)g, (las_v4i*)l, 0, 0);
#else
  *(half8*)l = *(const half8*)g;  // synchronous fallback
#endif
}

// ---------------------------------------------------------------- prep kernels
__global__ __launch_bounds__(256) void cvt_x(const float* __restrict__ x,
                                             _Float16* __restrict__ xh, int n) {
  int i = (blockIdx.x * 256 + threadIdx.x) * 4;
  if (i < n) {
    f4v v = *(const f4v*)(x + i);
    *(half4*)(xh + i) = __builtin_convertvector(v, half4);
  }
}

// Wt[w][n][k] = (f16) W_w[k][n]   (transpose so B-fragments read contiguous K)
__global__ __launch_bounds__(256) void cvt_w(const float* __restrict__ Wq,
                                             const float* __restrict__ Wk,
                                             const float* __restrict__ Wv,
                                             _Float16* __restrict__ wt) {
  int idx = blockIdx.x * 256 + threadIdx.x;
  if (idx >= 3 * NC * ND) return;
  int w = idx / (NC * ND);
  int rem = idx - w * (NC * ND);
  int n = rem % ND;
  int k = rem / ND;
  const float* W = (w == 0) ? Wq : (w == 1) ? Wk : Wv;
  wt[(size_t)w * NC * ND + (size_t)n * NC + k] = (_Float16)W[rem];
}

// ---------------------------------------------------------------- projection
// grid.x = (B*T)/128, grid.y = 3 (0:Q, 1:K, 2:V-transposed), 256 thr = 8 waves
// W k-chunks are double-buffered in LDS via async DMA; shared by all 8 waves.
__global__ __launch_bounds__(256) void proj(const _Float16* __restrict__ xh,
                                            const _Float16* __restrict__ wt,
                                            _Float16* __restrict__ qh,
                                            _Float16* __restrict__ kh,
                                            _Float16* __restrict__ vth) {
  __shared__ _Float16 wbuf[2][32 * 128];  // [k-chunk buffer][n*32 + k]

  const int tid  = threadIdx.x;
  const int wave = tid >> 5;
  const int lane = tid & 31;
  const int lrow = lane & 15;
  const int lhi  = lane >> 4;
  const int which = blockIdx.y;
  const int r0 = blockIdx.x * 128 + wave * 16;  // row over B*T

  const _Float16* W = wt + (size_t)which * (NC * ND);

  // staging: thread covers row n = tid/2, halfs wk..wk+15 of each 32-chunk
  const int wn = tid >> 1;
  const int wk = (tid & 1) * 16;
  const _Float16* wsrc = W + (size_t)wn * NC + wk;
  cp16_async(wsrc,     &wbuf[0][wn * 32 + wk]);
  cp16_async(wsrc + 8, &wbuf[0][wn * 32 + wk + 8]);

  f8v acc[8];
  #pragma unroll
  for (int nt = 0; nt < 8; ++nt)
    #pragma unroll
    for (int r = 0; r < 8; ++r) acc[nt][r] = 0.0f;

  const _Float16* arow = xh + (size_t)(r0 + lrow) * NC + lhi * 8;
  int cur = 0;
  for (int k0 = 0; k0 < NC; k0 += 32) {
    const bool hasnext = (k0 + 32 < NC);
    if (hasnext) {
      cp16_async(wsrc + k0 + 32,     &wbuf[1 - cur][wn * 32 + wk]);
      cp16_async(wsrc + k0 + 32 + 8, &wbuf[1 - cur][wn * 32 + wk + 8]);
      asm volatile("s_wait_asynccnt 0x2" ::: "memory");
    } else {
      asm volatile("s_wait_asynccnt 0x0" ::: "memory");
    }
    __syncthreads();

    half8 a0 = *(const half8*)(arow + k0);
    half8 a1 = *(const half8*)(arow + k0 + 16);
    half16 a = cat8(a0, a1);

    half16 bf[8];
    #pragma unroll
    for (int nt = 0; nt < 8; ++nt) {
      const _Float16* brow = &wbuf[cur][(nt * 16 + lrow) * 32 + lhi * 8];
      bf[nt] = cat8(*(const half8*)brow, *(const half8*)(brow + 16));
    }
    #pragma unroll
    for (int nt = 0; nt < 8; ++nt) acc[nt] = wmma16(a, bf[nt], acc[nt]);

    __syncthreads();
    if (hasnext) cur ^= 1;
  }

  const int b = r0 >> 11;  // / NT
  const int t = r0 & (NT - 1);
  if (which < 2) {
    _Float16* out = (which == 0) ? qh : kh;
    #pragma unroll
    for (int nt = 0; nt < 8; ++nt)
      #pragma unroll
      for (int r = 0; r < 8; ++r) {
        int row = r0 + r + 8 * lhi;
        out[(size_t)row * ND + nt * 16 + lrow] = (_Float16)acc[nt][r];
      }
  } else {
    #pragma unroll
    for (int nt = 0; nt < 8; ++nt)
      #pragma unroll
      for (int r = 0; r < 8; ++r) {
        int trow = t + r + 8 * lhi;
        vth[(size_t)b * ND * NT + (size_t)(nt * 16 + lrow) * NT + trow] =
            (_Float16)acc[nt][r];
      }
  }
}

// ---------------------------------------------------------------- flash attn
// grid.x = B * (T/128) = 256; 256 thr = 8 waves, each wave: 16 queries.
// K/V tiles double-buffered in LDS via async DMA (8 b128 asyncs / thread / tile).
__device__ __forceinline__ void stage_kv(const _Float16* ks, const _Float16* vs,
                                         _Float16* kt, _Float16* vt, int tid) {
  #pragma unroll
  for (int c = 0; c < 4; ++c) {
    int e = tid + 256 * c;
    cp16_async(ks + (size_t)e * 8, kt + e * 8);
  }
  #pragma unroll
  for (int c = 0; c < 4; ++c) {
    int e = tid + 256 * c;
    int d = e >> 3, c8 = (e & 7) * 8;
    cp16_async(vs + (size_t)d * NT + c8, vt + d * 64 + c8);
  }
}

__global__ __launch_bounds__(256) void flash(const _Float16* __restrict__ qh,
                                             const _Float16* __restrict__ kh,
                                             const _Float16* __restrict__ vth,
                                             float* __restrict__ out) {
  __shared__ _Float16 ktile[2][64 * 128];  // [buf][key][d]
  __shared__ _Float16 vtile[2][128 * 64];  // [buf][d][key]
  __shared__ _Float16 ptile[8 * 16 * 64];

  const int tid  = threadIdx.x;
  const int wave = tid >> 5;
  const int lane = tid & 31;
  const int lrow = lane & 15;
  const int lhi  = lane >> 4;
  const int b  = blockIdx.x >> 4;
  const int qb = blockIdx.x & 15;
  const int q0w = qb * 128 + wave * 16;          // wave's first query (in batch)
  const float scale = 0.08838834764831845f;      // 1/sqrt(128)

  // Q A-fragments resident for the whole key loop (16 rows x 128 d)
  half16 qa[4];
  const _Float16* qrow = qh + ((size_t)b * NT + q0w + lrow) * ND + lhi * 8;
  #pragma unroll
  for (int kc = 0; kc < 4; ++kc) {
    half8 a0 = *(const half8*)(qrow + kc * 32);
    half8 a1 = *(const half8*)(qrow + kc * 32 + 16);
    qa[kc] = cat8(a0, a1);
  }

  f8v o[8];
  #pragma unroll
  for (int dt = 0; dt < 8; ++dt)
    #pragma unroll
    for (int r = 0; r < 8; ++r) o[dt][r] = 0.0f;
  float m_run[8], l_run[8];
  #pragma unroll
  for (int r = 0; r < 8; ++r) { m_run[r] = -1e30f; l_run[r] = 0.0f; }

  const _Float16* kbase = kh + (size_t)b * NT * ND;
  const _Float16* vbase = vth + (size_t)b * ND * NT;
  const int jmax = qb * 128 + 128;

  stage_kv(kbase, vbase, ktile[0], vtile[0], tid);  // j0 = 0 tile
  int cur = 0;
  for (int j0 = 0; j0 < jmax; j0 += 64) {
    const bool hasnext = (j0 + 64 < jmax);
    if (hasnext) {
      stage_kv(kbase + (size_t)(j0 + 64) * ND, vbase + (j0 + 64),
               ktile[1 - cur], vtile[1 - cur], tid);
      asm volatile("s_wait_asynccnt 0x8" ::: "memory");
    } else {
      asm volatile("s_wait_asynccnt 0x0" ::: "memory");
    }
    __syncthreads();

    if (j0 <= q0w + 15) {
      const _Float16* kt = ktile[cur];
      const _Float16* vt = vtile[cur];
      // ---- S = Q K^T for 64 keys -------------------------------------------
      f8v s[4];
      #pragma unroll
      for (int nt = 0; nt < 4; ++nt) {
        #pragma unroll
        for (int r = 0; r < 8; ++r) s[nt][r] = 0.0f;
        #pragma unroll
        for (int kc = 0; kc < 4; ++kc) {
          const _Float16* brow = kt + (nt * 16 + lrow) * 128 + kc * 32 + lhi * 8;
          half8 b0 = *(const half8*)brow;
          half8 b1 = *(const half8*)(brow + 16);
          s[nt] = wmma16(qa[kc], cat8(b0, b1), s[nt]);
        }
      }
      // scale + causal mask (only blocks touching the diagonal)
      const bool diag = (j0 + 64 > q0w);
      #pragma unroll
      for (int nt = 0; nt < 4; ++nt)
        #pragma unroll
        for (int r = 0; r < 8; ++r) {
          float v = s[nt][r] * scale;
          if (diag) {
            int key = j0 + nt * 16 + lrow;
            int qq  = q0w + r + 8 * lhi;
            if (key > qq) v = -1e30f;
          }
          s[nt][r] = v;
        }
      // ---- online softmax ---------------------------------------------------
      float alpha[8];
      #pragma unroll
      for (int r = 0; r < 8; ++r) {
        float rm = s[0][r];
        rm = fmaxf(rm, s[1][r]); rm = fmaxf(rm, s[2][r]); rm = fmaxf(rm, s[3][r]);
        #pragma unroll
        for (int m = 1; m < 16; m <<= 1) rm = fmaxf(rm, __shfl_xor(rm, m, 32));
        float mnew = fmaxf(m_run[r], rm);
        alpha[r] = __expf(m_run[r] - mnew);
        m_run[r] = mnew;
      }
      #pragma unroll
      for (int r = 0; r < 8; ++r) {
        float acc = 0.0f;
        #pragma unroll
        for (int nt = 0; nt < 4; ++nt) {
          float p = __expf(s[nt][r] - m_run[r]);
          s[nt][r] = p;
          acc += p;
        }
        #pragma unroll
        for (int m = 1; m < 16; m <<= 1) acc += __shfl_xor(acc, m, 32);
        l_run[r] = l_run[r] * alpha[r] + acc;
      }
      #pragma unroll
      for (int dt = 0; dt < 8; ++dt)
        #pragma unroll
        for (int r = 0; r < 8; ++r) o[dt][r] *= alpha[r];

      // ---- P -> LDS, re-layout as A-fragment -------------------------------
      _Float16* pw = ptile + wave * (16 * 64);
      #pragma unroll
      for (int nt = 0; nt < 4; ++nt)
        #pragma unroll
        for (int r = 0; r < 8; ++r)
          pw[(r + 8 * lhi) * 64 + nt * 16 + lrow] = (_Float16)s[nt][r];
      asm volatile("s_wait_dscnt 0" ::: "memory");

      half16 pa[2];
      const _Float16* par = pw + lrow * 64 + lhi * 8;
      #pragma unroll
      for (int kc = 0; kc < 2; ++kc) {
        half8 a0 = *(const half8*)(par + kc * 32);
        half8 a1 = *(const half8*)(par + kc * 32 + 16);
        pa[kc] = cat8(a0, a1);
      }
      // ---- O += P V ---------------------------------------------------------
      #pragma unroll
      for (int dt = 0; dt < 8; ++dt)
        #pragma unroll
        for (int kc = 0; kc < 2; ++kc) {
          const _Float16* brow = vt + (dt * 16 + lrow) * 64 + kc * 32 + lhi * 8;
          half8 b0 = *(const half8*)brow;
          half8 b1 = *(const half8*)(brow + 16);
          o[dt] = wmma16(pa[kc], cat8(b0, b1), o[dt]);
        }
    }
    __syncthreads();
    if (hasnext) cur ^= 1;
  }

  // ---- normalize + store f32 ----------------------------------------------
  float inv[8];
  #pragma unroll
  for (int r = 0; r < 8; ++r) inv[r] = 1.0f / l_run[r];
  float* orow = out + ((size_t)b * NT + q0w) * ND;
  #pragma unroll
  for (int dt = 0; dt < 8; ++dt)
    #pragma unroll
    for (int r = 0; r < 8; ++r)
      orow[(size_t)(r + 8 * lhi) * ND + dt * 16 + lrow] = o[dt][r] * inv[r];
}

// ---------------------------------------------------------------- launcher
extern "C" void kernel_launch(void* const* d_in, const int* in_sizes, int n_in,
                              void* d_out, int out_size, void* d_ws, size_t ws_size,
                              hipStream_t stream) {
  const float* x  = (const float*)d_in[0];
  const float* Wq = (const float*)d_in[1];
  const float* Wk = (const float*)d_in[2];
  const float* Wv = (const float*)d_in[3];
  float* out = (float*)d_out;

  char* ws = (char*)d_ws;
  const size_t XH = 0;
  const size_t WT = XH + (size_t)NB * NT * NC * 2;   // 64 MB
  const size_t QH = WT + (size_t)3 * NC * ND * 2;    // +0.75 MB
  const size_t KH = QH + (size_t)NB * NT * ND * 2;   // +8 MB
  const size_t VT = KH + (size_t)NB * NT * ND * 2;   // +8 MB
  _Float16* xh  = (_Float16*)(ws + XH);
  _Float16* wt  = (_Float16*)(ws + WT);
  _Float16* qh  = (_Float16*)(ws + QH);
  _Float16* kh  = (_Float16*)(ws + KH);
  _Float16* vth = (_Float16*)(ws + VT);

  const int nx = NB * NT * NC;
  cvt_x<<<nx / (256 * 4), 256, 0, stream>>>(x, xh, nx);
  cvt_w<<<(3 * NC * ND + 255) / 256, 256, 0, stream>>>(Wq, Wk, Wv, wt);
  proj<<<dim3((NB * NT) / 128, 3), 256, 0, stream>>>(xh, wt, qh, kh, vth);
  flash<<<NB * (NT / 128), 256, 0, stream>>>(qh, kh, vth, out);
}
